// GATVar_29231547416618
// MI455X (gfx1250) — matrix operand order repaired and checked by reading, not compiled, not run on hardware
//
#include <hip/hip_runtime.h>

#define IN_FEATS 256
#define HIDDEN   64
#define HEADS    8
#define OUT_FEATS 40
#define NEG_SLOPE 0.2f

#define BK      64                 // K-slice staged in LDS (all K are multiples of 64)
#define BPITCH  (BK + 8)           // bf16 elements; row stride 144B = 36 dwords -> 16 distinct banks
                                   // and 16B-aligned rows for ds b128 reads

typedef __attribute__((ext_vector_type(16))) __bf16 v16bf;
typedef __attribute__((ext_vector_type(8)))  float  v8f;

union FragBF16 {                   // 32 bytes: one WMMA 16-bf16 operand fragment
  v16bf v;
  uint4 q[2];
};

// ---------------- helpers ----------------
__device__ __forceinline__ unsigned f2key(float f) {
  unsigned u = __float_as_uint(f);
  return (u & 0x80000000u) ? ~u : (u | 0x80000000u);   // monotonic order-preserving key
}
__device__ __forceinline__ float key2f(unsigned k) {
  return (k & 0x80000000u) ? __uint_as_float(k & 0x7FFFFFFFu) : __uint_as_float(~k);
}
__device__ __forceinline__ float lrelu(float v) { return v > 0.f ? v : NEG_SLOPE * v; }

// ---------------- fp32 -> bf16 (row-major copy) ----------------
__global__ __launch_bounds__(256) void k_f32_to_bf16(const float* __restrict__ in,
                                                     __bf16* __restrict__ out, int n) {
  int i = blockIdx.x * blockDim.x + threadIdx.x;
  if (i < n) out[i] = (__bf16)in[i];
}

// ---------------- fp32 [K,N] -> bf16 transposed [N,K] ----------------
__global__ __launch_bounds__(256) void k_f32_to_bf16_T(const float* __restrict__ in,
                                                       __bf16* __restrict__ out, int K, int N) {
  int i = blockIdx.x * blockDim.x + threadIdx.x;
  if (i >= K * N) return;
  int k = i / N, n = i - k * N;
  out[(size_t)n * K + k] = (__bf16)in[i];
}

// ---------------- WMMA GEMM: C[M,N] = A[M,K] * BT[N,K]^T, bf16 in, f32 out ----------------
// Block = 8 waves sharing one 64-column N-group; wave w owns M-tile (mblk*8 + w).
// B panel staged in LDS per 64-wide K-slice, shared by all 8 waves.
// All operand fragments for a k-slice are preloaded into registers before the WMMA
// burst so DS/VMEM latency overlaps the matrix pipe (graduated s_wait counts).
// Fragment layouts per CDNA5 ISA 7.12.2 (wave32):
//   A (16x32): lanes 0-15 row M=lane, elems 0-7 -> K 0-7, elems 8-15 -> K 16-23; hi lanes +8.
//   B (32x16): lanes 0-15 col N=lane, elems 0-7 -> K 0-7, elems 8-15 -> K 16-23; hi lanes +8.
__global__ __launch_bounds__(256) void k_wmma_gemm(const __bf16* __restrict__ A,
                                                   const __bf16* __restrict__ BT,
                                                   float* __restrict__ C,
                                                   int M, int N, int K,
                                                   int groupsN, int tilesM) {
  __shared__ __bf16 Bs[64 * BPITCH];          // 9216 bytes

  const int blk  = blockIdx.x;
  const int mblk = blk / groupsN;
  const int tg   = blk - mblk * groupsN;
  const int tid  = threadIdx.x;
  const int wave = tid >> 5;
  const int lane = tid & 31;

  const int tmRaw   = mblk * 8 + wave;
  const bool active = tmRaw < tilesM;
  const int tm = active ? tmRaw : (tilesM - 1);

  int row = tm * 16 + (lane & 15);
  if (row >= M) row = M - 1;                  // M multiple of 16 here; safety clamp
  const int kA = (lane >= 16) ? 8 : 0;
  const int kB = (lane >= 16) ? 8 : 0;
  const __bf16* Arow = A + (size_t)row * K;

  v8f acc[4] = {};
  const int chunksPerRow = BK / 8;            // uint4 (8 bf16) chunks per row slice
  const int totalChunks  = 64 * chunksPerRow;

  for (int k0 = 0; k0 < K; k0 += BK) {
    // ---- cooperative stage: BT[tg*64 .. +63][k0 .. k0+BK) -> LDS ----
    for (int idx = tid; idx < totalChunks; idx += 256) {
      int r  = idx / chunksPerRow;
      int cc = idx - r * chunksPerRow;
      int gcol = tg * 64 + r;
      if (gcol >= N) gcol = N - 1;            // clamp; OOB results discarded at store
      uint4 v = *(const uint4*)(BT + (size_t)gcol * K + k0 + cc * 8);
      *(uint4*)(&Bs[r * BPITCH + cc * 8]) = v;
    }
    __syncthreads();

    // ---- preload both A fragments of this slice (VMEM in flight) ----
    FragBF16 fa[2];
#pragma unroll
    for (int s = 0; s < 2; ++s) {
      fa[s].q[0] = *(const uint4*)(Arow + k0 + s * 32 + kA);        // K 0-7  (lane half)
      fa[s].q[1] = *(const uint4*)(Arow + k0 + s * 32 + kA + 16);   // K 16-23
    }

    // ---- 2 k-steps x 4 N-subtiles; B frags preloaded before the WMMA burst ----
#pragma unroll
    for (int kk = 0; kk < BK; kk += 32) {
      FragBF16 fb[4];
#pragma unroll
      for (int t = 0; t < 4; ++t) {
        const __bf16* bp = &Bs[(t * 16 + (lane & 15)) * BPITCH + kk + kB];
        fb[t].q[0] = *(const uint4*)(bp);                           // K kB..kB+7
        fb[t].q[1] = *(const uint4*)(bp + 16);                      // K kB+16..kB+23
      }
#pragma unroll
      for (int t = 0; t < 4; ++t) {
        acc[t] = __builtin_amdgcn_wmma_f32_16x16x32_bf16(false, fa[kk / 32].v, false, fb[t].v,
                                                         (short)0, acc[t], false, false);
      }
    }
    __syncthreads();
  }

  if (!active) return;
  const int rbase = tm * 16 + ((lane >= 16) ? 8 : 0);
#pragma unroll
  for (int t = 0; t < 4; ++t) {
    int col = tg * 64 + t * 16 + (lane & 15);
    if (col >= N) continue;
#pragma unroll
    for (int r = 0; r < 8; ++r) {
      int rr = rbase + r;
      if (rr < M) C[(size_t)rr * N + col] = acc[t][r];
    }
  }
}

// ---------------- attention score dots: a_src[n,h] = <xl[n,h,:], att_src[h,:]> ----------------
__global__ __launch_bounds__(256) void k_att(const float* __restrict__ xl,
                                             const float* __restrict__ att_s,
                                             const float* __restrict__ att_d,
                                             float* __restrict__ asrc, float* __restrict__ adst,
                                             int N, int H, int C) {
  int t = blockIdx.x * blockDim.x + threadIdx.x;
  if (t >= N * H) return;
  int n = t / H, h = t - n * H;
  const float* xr = xl + (size_t)n * H * C + (size_t)h * C;
  const float* ps = att_s + (size_t)h * C;
  const float* pd = att_d + (size_t)h * C;
  float s = 0.f, d = 0.f;
  for (int c = 0; c < C; ++c) { s += xr[c] * ps[c]; d += xr[c] * pd[c]; }
  asrc[t] = s;
  adst[t] = d;
}

// ---------------- per-edge segment max (uint-key atomicMax) ----------------
__global__ __launch_bounds__(256) void k_edge_max(const long long* __restrict__ ei, int E, int N, int H,
                                                  const float* __restrict__ asrc,
                                                  const float* __restrict__ adst,
                                                  unsigned* __restrict__ key) {
  int t = blockIdx.x * blockDim.x + threadIdx.x;
  int ET = E + N;
  if (t >= ET * H) return;
  int e = t / H, h = t - e * H;
  int s, d;
  if (e < E) { s = (int)ei[e]; d = (int)ei[E + e]; } else { s = d = e - E; }
  float v = lrelu(asrc[s * H + h] + adst[d * H + h]);
  atomicMax(&key[d * H + h], f2key(v));
}

// ---------------- per-edge exp + segment sum ----------------
__global__ __launch_bounds__(256) void k_edge_exp(const long long* __restrict__ ei, int E, int N, int H,
                                                  const float* __restrict__ asrc,
                                                  const float* __restrict__ adst,
                                                  const unsigned* __restrict__ key,
                                                  float* __restrict__ ee, float* __restrict__ den) {
  int t = blockIdx.x * blockDim.x + threadIdx.x;
  int ET = E + N;
  if (t >= ET * H) return;
  int e = t / H, h = t - e * H;
  int s, d;
  if (e < E) { s = (int)ei[e]; d = (int)ei[E + e]; } else { s = d = e - E; }
  float v = lrelu(asrc[s * H + h] + adst[d * H + h]);
  float x = __expf(v - key2f(key[d * H + h]));
  ee[t] = x;
  atomicAdd(&den[d * H + h], x);
}

// ---------------- per-(edge,head,channel) scatter aggregation ----------------
__global__ __launch_bounds__(256) void k_edge_agg(const long long* __restrict__ ei, int E, int N,
                                                  int H, int C,
                                                  const float* __restrict__ xl,
                                                  const float* __restrict__ ee,
                                                  const float* __restrict__ den,
                                                  float* __restrict__ agg) {
  long long t = (long long)blockIdx.x * blockDim.x + threadIdx.x;
  int HC = H * C;
  long long total = (long long)(E + N) * HC;
  if (t >= total) return;
  int e = (int)(t / HC);
  int rem = (int)(t - (long long)e * HC);
  int h = rem / C;
  int s, d;
  if (e < E) { s = (int)ei[e]; d = (int)ei[E + e]; } else { s = d = e - E; }
  float alpha = ee[(size_t)e * H + h] / den[d * H + h];
  atomicAdd(&agg[(size_t)d * HC + rem], alpha * xl[(size_t)s * HC + rem]);
}

// ---------------- layer1 finalize: elu(agg + bias) -> bf16 ----------------
__global__ __launch_bounds__(256) void k_elu_bias_bf16(const float* __restrict__ agg,
                                                       const float* __restrict__ bias,
                                                       __bf16* __restrict__ out, int N, int F) {
  int i = blockIdx.x * blockDim.x + threadIdx.x;
  if (i >= N * F) return;
  float v = agg[i] + bias[i % F];
  v = v > 0.f ? v : (__expf(v) - 1.f);
  out[i] = (__bf16)v;
}

// ---------------- layer2 finalize: emb = agg + bias (f32 out + bf16 copy) ----------------
__global__ __launch_bounds__(256) void k_bias_emb(const float* __restrict__ agg,
                                                  const float* __restrict__ bias,
                                                  float* __restrict__ emb,
                                                  __bf16* __restrict__ embbf, int N, int F) {
  int i = blockIdx.x * blockDim.x + threadIdx.x;
  if (i >= N * F) return;
  float v = agg[i] + bias[i % F];
  emb[i] = v;
  embbf[i] = (__bf16)v;
}

// ---------------- head: bias + softmax + argmax ----------------
__global__ __launch_bounds__(256) void k_head(float* __restrict__ logits,
                                              const float* __restrict__ bc,
                                              float* __restrict__ soft,
                                              float* __restrict__ hard, int N) {
  int n = blockIdx.x * blockDim.x + threadIdx.x;
  if (n >= N) return;
  float l[OUT_FEATS], ex[OUT_FEATS];
  float mx = -3.4e38f;
  for (int j = 0; j < OUT_FEATS; ++j) {
    float v = logits[(size_t)n * OUT_FEATS + j] + bc[j];
    l[j] = v;
    mx = fmaxf(mx, v);
  }
  float sum = 0.f;
  int arg = 0;
  float best = l[0];
  for (int j = 0; j < OUT_FEATS; ++j) {
    float v = __expf(l[j] - mx);
    ex[j] = v;
    sum += v;
    if (l[j] > best) { best = l[j]; arg = j; }
  }
  float inv = 1.f / sum;
  for (int j = 0; j < OUT_FEATS; ++j) {
    logits[(size_t)n * OUT_FEATS + j] = l[j];
    soft[(size_t)n * OUT_FEATS + j] = ex[j] * inv;
  }
  hard[n] = (float)arg;
}

// ---------------- launch ----------------
extern "C" void kernel_launch(void* const* d_in, const int* in_sizes, int n_in,
                              void* d_out, int out_size, void* d_ws, size_t ws_size,
                              hipStream_t stream) {
  (void)n_in; (void)out_size; (void)ws_size;
  const float*     x   = (const float*)d_in[0];
  const long long* ei  = (const long long*)d_in[1];     // int64 edge_index
  const float*     W1  = (const float*)d_in[2];
  const float*     as1 = (const float*)d_in[3];
  const float*     ad1 = (const float*)d_in[4];
  const float*     b1  = (const float*)d_in[5];
  const float*     W2  = (const float*)d_in[6];
  const float*     as2 = (const float*)d_in[7];
  const float*     ad2 = (const float*)d_in[8];
  const float*     b2  = (const float*)d_in[9];
  const float*     Wc  = (const float*)d_in[10];
  const float*     bc  = (const float*)d_in[11];

  const int N  = in_sizes[0] / IN_FEATS;
  const int E  = in_sizes[1] / 2;
  const int ET = E + N;                        // + self loops
  const int F1 = HEADS * HIDDEN;               // 512

  char* ws = (char*)d_ws;
  size_t off = 0;
  auto alloc = [&](size_t bytes) {
    char* p = ws + off;
    off += (bytes + 255) & ~(size_t)255;
    return p;
  };
  __bf16*   xbf   = (__bf16*)  alloc((size_t)N * IN_FEATS * 2);
  __bf16*   w1t   = (__bf16*)  alloc((size_t)IN_FEATS * F1 * 2);      // [F1, IN_FEATS]
  __bf16*   w2t   = (__bf16*)  alloc((size_t)F1 * HIDDEN * 2);        // [HIDDEN, F1]
  __bf16*   wct   = (__bf16*)  alloc((size_t)HIDDEN * OUT_FEATS * 2); // [OUT_FEATS, HIDDEN]
  float*    xl1   = (float*)   alloc((size_t)N * F1 * 4);
  float*    asrc1 = (float*)   alloc((size_t)N * HEADS * 4);
  float*    adst1 = (float*)   alloc((size_t)N * HEADS * 4);
  unsigned* key1  = (unsigned*)alloc((size_t)N * HEADS * 4);
  float*    den1  = (float*)   alloc((size_t)N * HEADS * 4);
  float*    ee1   = (float*)   alloc((size_t)ET * HEADS * 4);
  float*    agg1  = (float*)   alloc((size_t)N * F1 * 4);
  __bf16*   h1bf  = (__bf16*)  alloc((size_t)N * F1 * 2);
  float*    xl2   = (float*)   alloc((size_t)N * HIDDEN * 4);
  float*    asrc2 = (float*)   alloc((size_t)N * 4);
  float*    adst2 = (float*)   alloc((size_t)N * 4);
  unsigned* key2  = (unsigned*)alloc((size_t)N * 4);
  float*    den2  = (float*)   alloc((size_t)N * 4);
  float*    ee2   = (float*)   alloc((size_t)ET * 4);
  float*    agg2  = (float*)   alloc((size_t)N * HIDDEN * 4);
  __bf16*   embbf = (__bf16*)  alloc((size_t)N * HIDDEN * 2);

  float* out    = (float*)d_out;
  float* logits = out;                                  // [N,40]
  float* emb    = out + (size_t)N * OUT_FEATS;          // [N,64]
  float* soft   = emb + (size_t)N * HIDDEN;             // [N,40]
  float* hard   = soft + (size_t)N * OUT_FEATS;         // [N]

  auto grid = [](long long n) { return dim3((unsigned)((n + 255) / 256)); };

  // zero accumulators; key=0 decodes below -inf for the max-key trick
  hipMemsetAsync(key1, 0, (size_t)N * HEADS * 4, stream);
  hipMemsetAsync(den1, 0, (size_t)N * HEADS * 4, stream);
  hipMemsetAsync(agg1, 0, (size_t)N * F1 * 4, stream);
  hipMemsetAsync(key2, 0, (size_t)N * 4, stream);
  hipMemsetAsync(den2, 0, (size_t)N * 4, stream);
  hipMemsetAsync(agg2, 0, (size_t)N * HIDDEN * 4, stream);

  // bf16 conversions: activations row-major, weights transposed for contiguous B fragments
  k_f32_to_bf16<<<grid((long long)N * IN_FEATS), 256, 0, stream>>>(x, xbf, N * IN_FEATS);
  k_f32_to_bf16_T<<<grid(IN_FEATS * F1), 256, 0, stream>>>(W1, w1t, IN_FEATS, F1);
  k_f32_to_bf16_T<<<grid(F1 * HIDDEN), 256, 0, stream>>>(W2, w2t, F1, HIDDEN);
  k_f32_to_bf16_T<<<grid(HIDDEN * OUT_FEATS), 256, 0, stream>>>(Wc, wct, HIDDEN, OUT_FEATS);

  // ---- layer 1: xl1 = x @ W1 ----
  {
    int tilesM = (N + 15) / 16;
    int groupsN = (F1 + 63) / 64;
    int mblocks = (tilesM + 7) / 8;
    k_wmma_gemm<<<dim3(mblocks * groupsN), 256, 0, stream>>>(xbf, w1t, xl1, N, F1, IN_FEATS,
                                                             groupsN, tilesM);
  }
  k_att<<<grid((long long)N * HEADS), 256, 0, stream>>>(xl1, as1, ad1, asrc1, adst1, N, HEADS, HIDDEN);
  k_edge_max<<<grid((long long)ET * HEADS), 256, 0, stream>>>(ei, E, N, HEADS, asrc1, adst1, key1);
  k_edge_exp<<<grid((long long)ET * HEADS), 256, 0, stream>>>(ei, E, N, HEADS, asrc1, adst1, key1, ee1, den1);
  k_edge_agg<<<grid((long long)ET * F1), 256, 0, stream>>>(ei, E, N, HEADS, HIDDEN, xl1, ee1, den1, agg1);
  k_elu_bias_bf16<<<grid((long long)N * F1), 256, 0, stream>>>(agg1, b1, h1bf, N, F1);

  // ---- layer 2 (H=1): xl2 = h1 @ W2 ----
  {
    int tilesM = (N + 15) / 16;
    int groupsN = (HIDDEN + 63) / 64;
    int mblocks = (tilesM + 7) / 8;
    k_wmma_gemm<<<dim3(mblocks * groupsN), 256, 0, stream>>>(h1bf, w2t, xl2, N, HIDDEN, F1,
                                                             groupsN, tilesM);
  }
  k_att<<<grid((long long)N), 256, 0, stream>>>(xl2, as2, ad2, asrc2, adst2, N, 1, HIDDEN);
  k_edge_max<<<grid((long long)ET), 256, 0, stream>>>(ei, E, N, 1, asrc2, adst2, key2);
  k_edge_exp<<<grid((long long)ET), 256, 0, stream>>>(ei, E, N, 1, asrc2, adst2, key2, ee2, den2);
  k_edge_agg<<<grid((long long)ET * HIDDEN), 256, 0, stream>>>(ei, E, N, 1, HIDDEN, xl2, ee2, den2, agg2);
  k_bias_emb<<<grid((long long)N * HIDDEN), 256, 0, stream>>>(agg2, b2, emb, embbf, N, HIDDEN);

  // ---- classifier: logits = emb @ Wc ----
  {
    int tilesM = (N + 15) / 16;
    int groupsN = (OUT_FEATS + 63) / 64;
    int mblocks = (tilesM + 7) / 8;
    k_wmma_gemm<<<dim3(mblocks * groupsN), 256, 0, stream>>>(embbf, wct, logits, N, OUT_FEATS, HIDDEN,
                                                             groupsN, tilesM);
  }
  k_head<<<grid((long long)N), 256, 0, stream>>>(logits, bc, soft, hard, N);
}
